// MultiHeadAttention_43946105372998
// MI455X (gfx1250) — matrix operand order repaired and checked by reading, not compiled, not run on hardware
//
#include <hip/hip_runtime.h>
#include <hip/hip_bf16.h>

// MHA forward B=2,S=2048,D=1024,H=16,HS=64,OUT=1024 on gfx1250 (wave32).
// Pre-converted bf16 operands; GEMMs use 128x128 block tiles (8 WMMA per wave
// per k-step), double-buffered async global->LDS staging; flash attention.

#define BB   2
#define SS   2048
#define DD   1024
#define HH   16
#define HSZ  64
#define NHID 1024          // H*HS
#define MTOT 4096          // B*S
#define OUTD 1024

typedef __bf16 bf16_t;
typedef __attribute__((ext_vector_type(16))) __bf16        v16bf;
typedef __attribute__((ext_vector_type(2)))  __bf16        v2bf;
typedef __attribute__((ext_vector_type(8)))  float         v8f;
typedef __attribute__((ext_vector_type(8)))  unsigned int  v8u;
typedef __attribute__((ext_vector_type(4)))  int           v4i;

// ---- fp32 -> bf16 (lowers to v_cvt_pk_bf16_f32; probe-confirmed) --------
static __device__ __forceinline__ unsigned short bf1(float f) {
  bf16_t b = (bf16_t)f;
  return __builtin_bit_cast(unsigned short, b);
}
static __device__ __forceinline__ unsigned pack_bf(float lo, float hi) {
  v2bf p;
  p[0] = (bf16_t)lo;
  p[1] = (bf16_t)hi;
  return __builtin_bit_cast(unsigned, p);
}

// ---- async global->LDS staging (GLOBAL_LOAD_ASYNC_TO_LDS_B128) ----------
#if defined(__has_builtin)
#if __has_builtin(__builtin_amdgcn_global_load_async_to_lds_b128)
#define HAVE_ASYNC_LDS 1
#endif
#endif

typedef __attribute__((address_space(1))) v4i* g_v4i_p;   // global v4i*
typedef __attribute__((address_space(3))) v4i* l_v4i_p;   // LDS v4i*

static __device__ __forceinline__ void stage16(const void* g, void* l) {
#if defined(HAVE_ASYNC_LDS)
  // flat LDS address low 32 bits == LDS byte offset (ISA 10.2 aperture rule)
  g_v4i_p gp = (g_v4i_p)(unsigned long long)(uintptr_t)g;
  l_v4i_p lp = (l_v4i_p)(unsigned)(uintptr_t)l;
  __builtin_amdgcn_global_load_async_to_lds_b128(gp, lp, 0, 0);
#else
  *(uint4*)l = *(const uint4*)g;
#endif
}
static __device__ __forceinline__ void stage_wait() {
#if defined(HAVE_ASYNC_LDS)
#if __has_builtin(__builtin_amdgcn_s_wait_asynccnt)
  __builtin_amdgcn_s_wait_asynccnt(0);
#else
  asm volatile("s_wait_asynccnt 0x0" ::: "memory");
#endif
#endif
}

// A-matrix (16x32 bf16) K-pair base for element pair v (0..7), lane half hf.
static __device__ __forceinline__ int a_kpair(int v, int hf) {
  return hf * 16 + 2 * v;
}

static __device__ __forceinline__ v8f wmma_bf16(v16bf a, v16bf b, v8f c) {
  return __builtin_amdgcn_wmma_f32_16x16x32_bf16(false, a, false, b, (short)0, c,
                                                 false, false);
}

// ---- fragment loaders (bf16 source; global or LDS pointers) --------------
// A frag; row already includes lane%16.
static __device__ __forceinline__ v16bf load_a_bf16(const bf16_t* M, int ld,
                                                    int row, int c0, int hf) {
  v8u a;
  const bf16_t* p = M + (size_t)row * ld + c0;
#pragma unroll
  for (int v = 0; v < 8; ++v)
    a[v] = *(const unsigned*)(p + a_kpair(v, hf));
  return __builtin_bit_cast(v16bf, a);
}
// B frag where B[kk][n] = M[rown][c0+kk]  (transposed source; pairs contiguous).
static __device__ __forceinline__ v16bf load_bT_bf16(const bf16_t* M, int ld,
                                                     int rown, int c0, int hf) {
  v8u a;
  const bf16_t* p = M + (size_t)rown * ld + c0 + hf * 16;
#pragma unroll
  for (int v = 0; v < 8; ++v)
    a[v] = *(const unsigned*)(p + 2 * v);
  return __builtin_bit_cast(v16bf, a);
}
// B frag where B[kk][n] = M[k0+kk][col]  (pairs stride ld).
static __device__ __forceinline__ v16bf load_b_bf16(const bf16_t* M, int ld,
                                                    int k0, int col, int hf) {
  v8u a;
  const bf16_t* p = M + (size_t)(k0 + hf * 16) * ld + col;
#pragma unroll
  for (int v = 0; v < 8; ++v) {
    const unsigned lo = __builtin_bit_cast(unsigned short, p[(size_t)(2 * v) * ld]);
    const unsigned hi = __builtin_bit_cast(unsigned short, p[(size_t)(2 * v + 1) * ld]);
    a[v] = (hi << 16) | lo;
  }
  return __builtin_bit_cast(v16bf, a);
}

// ---- kernel 0a: activations fp32 -> bf16 (one-time) ----------------------
// grid = (MTOT*DD/2048, 3), block = 256; 8 elems/thread
__global__ __launch_bounds__(256) void cvt_x_kernel(
    const float* __restrict__ q, const float* __restrict__ k,
    const float* __restrict__ v, bf16_t* __restrict__ xb) {
  const int sel = blockIdx.y;
  const float* src = (sel == 0) ? q : (sel == 1) ? k : v;
  bf16_t* dst = xb + (size_t)sel * MTOT * DD;
  const size_t i0 = ((size_t)blockIdx.x * 256 + threadIdx.x) * 8;
  const float4 f0 = *(const float4*)(src + i0);
  const float4 f1 = *(const float4*)(src + i0 + 4);
  uint4 o;
  o.x = pack_bf(f0.x, f0.y);  o.y = pack_bf(f0.z, f0.w);
  o.z = pack_bf(f1.x, f1.y);  o.w = pack_bf(f1.z, f1.w);
  *(uint4*)(dst + i0) = o;
}

// ---- kernel 0b: weights fp32 -> bf16 transposed (one-time) ---------------
// WT[n][k] = W[k][n]; grid = (1024*1024/1024, 4), block = 256; 4 elems/thread
__global__ __launch_bounds__(256) void cvt_w_kernel(
    const float* __restrict__ Wq, const float* __restrict__ Wk,
    const float* __restrict__ Wv, const float* __restrict__ Wo,
    bf16_t* __restrict__ WT) {
  const int sel = blockIdx.y;
  const float* W = (sel == 0) ? Wq : (sel == 1) ? Wk : (sel == 2) ? Wv : Wo;
  bf16_t* T = WT + (size_t)sel * DD * NHID;
  const int lin = (blockIdx.x * 256 + threadIdx.x) * 4;
  const int k = lin >> 10, n = lin & 1023;
  const float4 f = *(const float4*)(W + (size_t)k * NHID + n);
  T[(size_t)(n + 0) * DD + k] = __builtin_bit_cast(bf16_t, bf1(f.x));
  T[(size_t)(n + 1) * DD + k] = __builtin_bit_cast(bf16_t, bf1(f.y));
  T[(size_t)(n + 2) * DD + k] = __builtin_bit_cast(bf16_t, bf1(f.z));
  T[(size_t)(n + 3) * DD + k] = __builtin_bit_cast(bf16_t, bf1(f.w));
}

// ---- 128x128 block-tile GEMM staging: A 128x32 + BT 128x32 (bf16) --------
static __device__ __forceinline__ void stage_gemm128(
    const bf16_t* __restrict__ Arows, const bf16_t* __restrict__ BTrows,
    int k0, int t, unsigned short (*A)[32], unsigned short (*BT)[32]) {
#pragma unroll
  for (int i = 0; i < 2; ++i) {
    const int id = i * 256 + t;            // 0..511 chunks of 16B
    const int rr = id >> 2, cb = (id & 3) * 8;
    stage16(Arows + (size_t)rr * DD + k0 + cb, &A[rr][cb]);
    stage16(BTrows + (size_t)rr * DD + k0 + cb, &BT[rr][cb]);
  }
}

// ---- kernel 1: fused QKV projection (128x128 tiles, double-buffered) -----
// grid = (MTOT/128, 3*NHID/128), block = 256 (8 waves; wave owns 16x128 strip)
__global__ __launch_bounds__(256) void qkv_proj_kernel(
    const bf16_t* __restrict__ xb,   // [3][MTOT][DD] bf16
    const bf16_t* __restrict__ WT,   // [4][N][K] bf16 (q,k,v,o)
    const float* __restrict__ bq, const float* __restrict__ bk,
    const float* __restrict__ bv, bf16_t* __restrict__ qkv_ws) {
  __shared__ __align__(16) unsigned short A_lds[2][128][32];   // 16KB
  __shared__ __align__(16) unsigned short BT_lds[2][128][32];  // 16KB

  const int t = threadIdx.x;
  const int lane = t & 31, ln = lane & 15, hf = lane >> 4, w = t >> 5;
  const int m0 = blockIdx.x * 128;
  const int nblk0 = blockIdx.y * 128;            // block stays in one region
  const int sel = nblk0 >> 10;                   // 0=q, 1=k, 2=v
  const int colb0 = nblk0 & 1023;

  const bf16_t* Arows  = xb + (size_t)sel * MTOT * DD + (size_t)m0 * DD;
  const bf16_t* BTrows = WT + (size_t)sel * DD * NHID + (size_t)colb0 * DD;
  const float* bias = (sel == 0) ? bq : (sel == 1) ? bk : bv;
  bf16_t* outp = qkv_ws + (size_t)sel * MTOT * NHID;

  stage_gemm128(Arows, BTrows, 0, t, A_lds[0], BT_lds[0]);
  stage_wait();
  __syncthreads();

  v8f acc[8] = {};
#pragma unroll 1
  for (int it = 0; it < DD / 32; ++it) {
    const int cur = it & 1;
    if (it + 1 < DD / 32)  // prefetch next k-tile into the other buffer
      stage_gemm128(Arows, BTrows, (it + 1) * 32, t, A_lds[cur ^ 1],
                    BT_lds[cur ^ 1]);
    const v16bf a =
        load_a_bf16((const bf16_t*)&A_lds[cur][0][0], 32, w * 16 + ln, 0, hf);
#pragma unroll
    for (int j = 0; j < 8; ++j) {
      const v16bf b = load_bT_bf16((const bf16_t*)&BT_lds[cur][0][0], 32,
                                   j * 16 + ln, 0, hf);
      acc[j] = wmma_bf16(a, b, acc[j]);
    }
    stage_wait();
    __syncthreads();
  }
  const int mrow0 = m0 + w * 16;
#pragma unroll
  for (int j = 0; j < 8; ++j) {
    const int col = colb0 + j * 16 + ln;
    const float bia = bias[col];
#pragma unroll
    for (int r = 0; r < 8; ++r) {
      outp[(size_t)(mrow0 + r + 8 * hf) * NHID + col] =
          __builtin_bit_cast(bf16_t, bf1(acc[j][r] + bia));
    }
  }
}

// ---- kernel 2: flash attention, block-shared double-buffered K/V ---------
// grid = (B*H, (S/16)/8), block = 256 (8 waves, each a 16-row Q tile)
__global__ __launch_bounds__(256) void attn_kernel(
    const bf16_t* __restrict__ q_ws, const bf16_t* __restrict__ k_ws,
    const bf16_t* __restrict__ v_ws, const int* __restrict__ v_mask,
    bf16_t* __restrict__ attn_ws) {
  __shared__ __align__(16) unsigned short K_lds[2][32][64];   // 8KB
  __shared__ __align__(16) unsigned short V_lds[2][32][64];   // 8KB
  __shared__ __align__(16) unsigned short p_lds[8][16][32];   // 8KB (per-wave P)

  const int t = threadIdx.x;
  const int lane = t & 31, ln = lane & 15, hf = lane >> 4, w = t >> 5;
  const int bh = blockIdx.x;
  const int b = bh >> 4, head = bh & 15;
  const int qt = blockIdx.y * 8 + w;
  const int qrow0 = b * SS + qt * 16;
  const int c0 = head * HSZ;
  const int srow = t >> 3, scol = (t & 7) * 8;   // 16B chunk per thread

  const v16bf aq0 = load_a_bf16(q_ws, NHID, qrow0 + ln, c0, hf);
  const v16bf aq1 = load_a_bf16(q_ws, NHID, qrow0 + ln, c0 + 32, hf);

  v8f o0 = {}, o1 = {}, o2 = {}, o3 = {};
  float mrow[8], lrow[8];
#pragma unroll
  for (int r = 0; r < 8; ++r) { mrow[r] = -3.0e38f; lrow[r] = 0.0f; }

  stage16(k_ws + (size_t)(b * SS + srow) * NHID + c0 + scol, &K_lds[0][srow][scol]);
  stage16(v_ws + (size_t)(b * SS + srow) * NHID + c0 + scol, &V_lds[0][srow][scol]);
  stage_wait();
  __syncthreads();

#pragma unroll 1
  for (int kt = 0; kt < SS / 32; ++kt) {
    const int cur = kt & 1;
    const int key0 = kt * 32;
    if (kt + 1 < SS / 32) {  // prefetch next K/V tile
      const size_t nb = (size_t)(b * SS + key0 + 32 + srow) * NHID + c0 + scol;
      stage16(k_ws + nb, &K_lds[cur ^ 1][srow][scol]);
      stage16(v_ws + nb, &V_lds[cur ^ 1][srow][scol]);
    }
    const bf16_t* Kt = (const bf16_t*)&K_lds[cur][0][0];
    const bf16_t* Vt = (const bf16_t*)&V_lds[cur][0][0];

    v8f s0 = {}, s1 = {};
    s0 = wmma_bf16(aq0, load_bT_bf16(Kt, 64, 0 + ln, 0, hf), s0);
    s0 = wmma_bf16(aq1, load_bT_bf16(Kt, 64, 0 + ln, 32, hf), s0);
    s1 = wmma_bf16(aq0, load_bT_bf16(Kt, 64, 16 + ln, 0, hf), s1);
    s1 = wmma_bf16(aq1, load_bT_bf16(Kt, 64, 16 + ln, 32, hf), s1);

    const float msk0 = v_mask[b * SS + key0 + ln]      ? 0.0f : -1.0e12f;
    const float msk1 = v_mask[b * SS + key0 + 16 + ln] ? 0.0f : -1.0e12f;

    float p0[8], p1[8], alpha[8];
#pragma unroll
    for (int r = 0; r < 8; ++r) {
      p0[r] = s0[r] * 0.125f + msk0;   // 1/sqrt(64)
      p1[r] = s1[r] * 0.125f + msk1;
      float mx = fmaxf(p0[r], p1[r]);  // row max over 16-lane half group
      mx = fmaxf(mx, __shfl_xor(mx, 1, 16));
      mx = fmaxf(mx, __shfl_xor(mx, 2, 16));
      mx = fmaxf(mx, __shfl_xor(mx, 4, 16));
      mx = fmaxf(mx, __shfl_xor(mx, 8, 16));
      const float mnew = fmaxf(mrow[r], mx);
      alpha[r] = __expf(mrow[r] - mnew);
      mrow[r] = mnew;
      p0[r] = __expf(p0[r] - mnew);
      p1[r] = __expf(p1[r] - mnew);
      float sum = p0[r] + p1[r];
      sum += __shfl_xor(sum, 1, 16);
      sum += __shfl_xor(sum, 2, 16);
      sum += __shfl_xor(sum, 4, 16);
      sum += __shfl_xor(sum, 8, 16);
      lrow[r] = lrow[r] * alpha[r] + sum;
    }
#pragma unroll
    for (int r = 0; r < 8; ++r) {
      o0[r] *= alpha[r]; o1[r] *= alpha[r];
      o2[r] *= alpha[r]; o3[r] *= alpha[r];
    }
#pragma unroll
    for (int r = 0; r < 8; ++r) {
      p_lds[w][r + 8 * hf][ln]      = bf1(p0[r]);
      p_lds[w][r + 8 * hf][16 + ln] = bf1(p1[r]);
    }
    asm volatile("s_wait_dscnt 0x0" ::: "memory");
    v16bf pa;
    {
      v8u tt;
#pragma unroll
      for (int v = 0; v < 8; ++v)
        tt[v] = *(const unsigned*)&p_lds[w][ln][a_kpair(v, hf)];
      pa = __builtin_bit_cast(v16bf, tt);
    }
    o0 = wmma_bf16(pa, load_b_bf16(Vt, 64, 0,  0 + ln, hf), o0);
    o1 = wmma_bf16(pa, load_b_bf16(Vt, 64, 0, 16 + ln, hf), o1);
    o2 = wmma_bf16(pa, load_b_bf16(Vt, 64, 0, 32 + ln, hf), o2);
    o3 = wmma_bf16(pa, load_b_bf16(Vt, 64, 0, 48 + ln, hf), o3);

    stage_wait();
    __syncthreads();
  }
#pragma unroll
  for (int r = 0; r < 8; ++r) {
    const float inv = 1.0f / lrow[r];
    const int row = qrow0 + r + 8 * hf;
    bf16_t* op = attn_ws + (size_t)row * NHID + c0;
    op[ 0 + ln] = __builtin_bit_cast(bf16_t, bf1(o0[r] * inv));
    op[16 + ln] = __builtin_bit_cast(bf16_t, bf1(o1[r] * inv));
    op[32 + ln] = __builtin_bit_cast(bf16_t, bf1(o2[r] * inv));
    op[48 + ln] = __builtin_bit_cast(bf16_t, bf1(o3[r] * inv));
  }
}

// ---- kernel 3: output projection + bias + query-mask ---------------------
// grid = (MTOT/128, OUTD/128), block = 256
__global__ __launch_bounds__(256) void out_proj_kernel(
    const bf16_t* __restrict__ attn_ws, const bf16_t* __restrict__ WoT,
    const float* __restrict__ bo, const int* __restrict__ q_mask,
    float* __restrict__ out) {
  __shared__ __align__(16) unsigned short A_lds[2][128][32];
  __shared__ __align__(16) unsigned short BT_lds[2][128][32];

  const int t = threadIdx.x;
  const int lane = t & 31, ln = lane & 15, hf = lane >> 4, w = t >> 5;
  const int m0 = blockIdx.x * 128;
  const int colb0 = blockIdx.y * 128;

  const bf16_t* Arows  = attn_ws + (size_t)m0 * NHID;
  const bf16_t* BTrows = WoT + (size_t)colb0 * NHID;

  stage_gemm128(Arows, BTrows, 0, t, A_lds[0], BT_lds[0]);
  stage_wait();
  __syncthreads();

  v8f acc[8] = {};
#pragma unroll 1
  for (int it = 0; it < NHID / 32; ++it) {
    const int cur = it & 1;
    if (it + 1 < NHID / 32)
      stage_gemm128(Arows, BTrows, (it + 1) * 32, t, A_lds[cur ^ 1],
                    BT_lds[cur ^ 1]);
    const v16bf a =
        load_a_bf16((const bf16_t*)&A_lds[cur][0][0], 32, w * 16 + ln, 0, hf);
#pragma unroll
    for (int j = 0; j < 8; ++j) {
      const v16bf b = load_bT_bf16((const bf16_t*)&BT_lds[cur][0][0], 32,
                                   j * 16 + ln, 0, hf);
      acc[j] = wmma_bf16(a, b, acc[j]);
    }
    stage_wait();
    __syncthreads();
  }
  const int mrow0 = m0 + w * 16;
#pragma unroll
  for (int j = 0; j < 8; ++j) {
    const int col = colb0 + j * 16 + ln;
    const float bias = bo[col];
#pragma unroll
    for (int r = 0; r < 8; ++r) {
      const int row = mrow0 + r + 8 * hf;
      const float qm = (float)q_mask[row];
      out[(size_t)row * OUTD + col] = (acc[j][r] + bias) * qm;
    }
  }
}

// ---- host-side launch ----------------------------------------------------
extern "C" void kernel_launch(void* const* d_in, const int* in_sizes, int n_in,
                              void* d_out, int out_size, void* d_ws, size_t ws_size,
                              hipStream_t stream) {
  const float* query = (const float*)d_in[0];
  const float* key   = (const float*)d_in[1];
  const float* value = (const float*)d_in[2];
  const float* Wq = (const float*)d_in[3];  const float* bq = (const float*)d_in[4];
  const float* Wk = (const float*)d_in[5];  const float* bk = (const float*)d_in[6];
  const float* Wv = (const float*)d_in[7];  const float* bv = (const float*)d_in[8];
  const float* Wo = (const float*)d_in[9];  const float* bo = (const float*)d_in[10];
  const int* q_mask = (const int*)d_in[11];
  const int* v_mask = (const int*)d_in[12];
  float* out = (float*)d_out;

  // workspace layout (bf16): xb 24MB | WT 8MB | qkv 24MB | attn 8MB = 64MB
  bf16_t* xb   = (bf16_t*)d_ws;
  bf16_t* WT   = xb + (size_t)3 * MTOT * DD;
  bf16_t* qkv  = WT + (size_t)4 * DD * NHID;
  bf16_t* attn = qkv + (size_t)3 * MTOT * NHID;

  dim3 blk(256);
  cvt_x_kernel<<<dim3((MTOT * DD) / 2048, 3), blk, 0, stream>>>(
      query, key, value, xb);
  cvt_w_kernel<<<dim3((DD * NHID) / 1024, 4), blk, 0, stream>>>(
      Wq, Wk, Wv, Wo, WT);
  qkv_proj_kernel<<<dim3(MTOT / 128, (3 * NHID) / 128), blk, 0, stream>>>(
      xb, WT, bq, bk, bv, qkv);
  attn_kernel<<<dim3(BB * HH, (SS / 16) / 8), blk, 0, stream>>>(
      qkv, qkv + (size_t)MTOT * NHID, qkv + (size_t)2 * MTOT * NHID, v_mask, attn);
  out_proj_kernel<<<dim3(MTOT / 128, OUTD / 128), blk, 0, stream>>>(
      attn, WT + (size_t)3 * DD * NHID, bo, q_mask, out);
}
